// PairStr2Pair_59974923321411
// MI455X (gfx1250) — matrix operand order repaired and checked by reading, not compiled, not run on hardware
//
#include <hip/hip_runtime.h>

// ---------------------------------------------------------------------------
// Types matching the gfx1250 WMMA builtin signatures
// ---------------------------------------------------------------------------
typedef __attribute__((ext_vector_type(16))) __bf16   v16bf;
typedef __attribute__((ext_vector_type(8)))  __bf16   bf16x8;
typedef __attribute__((ext_vector_type(8)))  float    v8f;
typedef __attribute__((ext_vector_type(4)))  unsigned u32x4;

#define DEVINL __device__ __forceinline__

constexpr int    LDIM = 512;
constexpr size_t LL   = (size_t)LDIM * LDIM;   // 262144 rows

// ---------------------------------------------------------------------------
// Helpers
// ---------------------------------------------------------------------------
DEVINL __bf16 f2bf(float f) {                       // fp32 -> bf16 RNE
  unsigned u = __builtin_bit_cast(unsigned, f);
  u += 0x7fffu + ((u >> 16) & 1u);
  unsigned short h = (unsigned short)(u >> 16);
  return __builtin_bit_cast(__bf16, h);
}

// (a*512+b) -> (b*512+a): row<->col transpose of the 512x512 grid (involution)
DEVINL size_t maptr(int m) { return ((size_t)(m & 511) << 9) | (size_t)((unsigned)m >> 9); }

// Concatenate two 8-elem bf16 vectors into the 16-elem WMMA operand.
// Pure register concat: fragment VGPRs 0..3 = lo, 4..7 = hi.
DEVINL v16bf cat8(bf16x8 lo, bf16x8 hi) {
  return __builtin_shufflevector(lo, hi, 0, 1, 2, 3, 4, 5, 6, 7,
                                 8, 9, 10, 11, 12, 13, 14, 15);
}

// A fragment (16x32 bf16, MxK), K contiguous in memory (row stride rs elems).
// Layout: lane m=lane&15, half=lane>>4; element s -> K=16*half+s
// (contiguous, matching frag_b).  Two adjacent 16B chunks per lane.
DEVINL v16bf frag_a(const __bf16* rowbase, long rs, int lane) {
  const int m = lane & 15, half = lane >> 4;
  const __bf16* p = rowbase + (long)m * rs + 16 * half;
  bf16x8 lo = *(const bf16x8*)p;          // K 16h .. 16h+7
  bf16x8 hi = *(const bf16x8*)(p + 8);    // K 16h+8 .. 16h+15
  return cat8(lo, hi);
}

// B fragment (32x16, KxN) from [N][K]-major bf16: lane n=lane&15 needs
// K = 16*half .. 16*half+15 -> one contiguous, 32B-aligned 32-byte load.
DEVINL v16bf frag_b(const __bf16* rowbase, long rs, int lane) {
  const int n = lane & 15, half = lane >> 4;
  return *(const v16bf*)(rowbase + (long)n * rs + 16 * half);
}

// B fragment from [K][N]-major bf16 (K strided by rs) — used for the V matrix
DEVINL v16bf frag_b_t(const __bf16* base, long rs, int lane) {
  const int n = lane & 15, half = lane >> 4;
  const __bf16* p = base + (long)(16 * half) * rs + n;
  v16bf b;
#pragma unroll
  for (int e = 0; e < 16; ++e) b[e] = p[(long)e * rs];
  return b;
}

DEVINL v8f wmma_bf16(v16bf a, v16bf b, v8f c) {
  return __builtin_amdgcn_wmma_f32_16x16x32_bf16(false, a, false, b, (short)0, c,
                                                 false, false);
}

DEVINL float wred_sum(float v) {
#pragma unroll
  for (int o = 16; o > 0; o >>= 1) v += __shfl_xor(v, o, 32);
  return v;
}
DEVINL float wred_max(float v) {
#pragma unroll
  for (int o = 16; o > 0; o >>= 1) v = fmaxf(v, __shfl_xor(v, o, 32));
  return v;
}

// ---------------------------------------------------------------------------
// Weight prep: W (KxN fp32) -> WT (Npad x K bf16, row-major, scaled, zero-pad)
// Weights are tiny (<=128 KB) -> they stay resident in the 192 MB L2.
// ---------------------------------------------------------------------------
__global__ __launch_bounds__(256) void k_prep(const float* __restrict__ Wsrc,
                                              __bf16* __restrict__ WT, int K, int N,
                                              int Npad, float scale) {
  const int idx = blockIdx.x * 256 + threadIdx.x;
  if (idx >= Npad * K) return;
  const int n = idx / K, kk = idx % K;
  const float v = (n < N) ? Wsrc[(size_t)kk * N + n] * scale : 0.0f;
  WT[idx] = f2bf(v);
}

// ---------------------------------------------------------------------------
// LayerNorm over C channels -> bf16.  One wave per row (wave32).
// TR: read the input with (i,j) transposed (row-attention swapaxes).
// ---------------------------------------------------------------------------
template <int C, bool TR>
__global__ __launch_bounds__(256) void k_ln(const float* __restrict__ x,
                                            const float* __restrict__ g,
                                            const float* __restrict__ b,
                                            __bf16* __restrict__ out) {
  const int row  = blockIdx.x * 8 + (threadIdx.x >> 5);
  const int lane = threadIdx.x & 31;
  const size_t ir = TR ? maptr(row) : (size_t)row;
  const float* xp = x + ir * C;
  constexpr int T = C / 32;
  float v[T];
  float s = 0.f;
#pragma unroll
  for (int t = 0; t < T; ++t) { v[t] = xp[t * 32 + lane]; s += v[t]; }
  s = wred_sum(s);
  const float mean = s * (1.0f / C);
  float vs = 0.f;
#pragma unroll
  for (int t = 0; t < T; ++t) { v[t] -= mean; vs += v[t] * v[t]; }
  vs = wred_sum(vs);
  const float inv = rsqrtf(vs * (1.0f / C) + 1e-5f);
#pragma unroll
  for (int t = 0; t < T; ++t) {
    const int c = t * 32 + lane;
    out[(size_t)row * C + c] = f2bf(v[t] * inv * g[c] + b[c]);
  }
}

// ---------------------------------------------------------------------------
// Generic WMMA GEMM: C(M=LL,NVALID) = A(M,K)bf16 x WT(NpadxK)bf16 [+bias][epi]
// Block = 128 threads (4 waves). 16-row A tile staged once in LDS (b128 ds
// traffic), each wave owns one 16-col tile; weights come straight from L2.
// ---------------------------------------------------------------------------
enum { EPI_NONE = 0, EPI_SIG = 1, EPI_RELU = 2, EPI_RES = 3 };

template <int K, int NPAD, int NVALID, int EPI, bool OBF, bool TR>
__global__ __launch_bounds__(128) void k_gemm(const __bf16* __restrict__ A,
                                              const __bf16* __restrict__ WT,
                                              const float* __restrict__ bias,
                                              const float* __restrict__ resid,
                                              void* __restrict__ outp) {
  constexpr int KC = K / 8;  // 16-byte chunks per row
  __shared__ __align__(16) __bf16 lAs[16 * K];
  const int m0   = blockIdx.x * 16;
  const int wave = threadIdx.x >> 5;
  const int lane = threadIdx.x & 31;

  // Stage the 16xK A tile via b128 global loads + b128 ds stores.
  {
    u32x4* dst = (u32x4*)lAs;
    const u32x4* src = (const u32x4*)A;
    for (int idx = threadIdx.x; idx < 16 * KC; idx += 128) {
      const int r = idx / KC, c = idx % KC;
      dst[idx] = src[(size_t)(m0 + r) * KC + c];
    }
  }
  __syncthreads();

  const int n0 = (blockIdx.y * 4 + wave) * 16;
  if (n0 >= NPAD) return;  // wave-uniform: EXEC stays all-ones for WMMA

  v8f acc = {};
#pragma unroll
  for (int kt = 0; kt < K / 32; ++kt) {
    v16bf a = frag_a((const __bf16*)lAs + kt * 32, K, lane);
    v16bf b = frag_b(WT + (size_t)n0 * K + kt * 32, K, lane);
    acc = wmma_bf16(a, b, acc);
  }

  const int n    = n0 + (lane & 15);
  const int half = lane >> 4;
  if (n >= NVALID) return;
  const float bv = bias ? bias[n] : 0.0f;
#pragma unroll
  for (int r = 0; r < 8; ++r) {
    const int m    = m0 + half * 8 + r;
    const size_t o = TR ? maptr(m) : (size_t)m;
    const size_t idx = o * (size_t)NVALID + n;
    float x = acc[r] + bv;
    if (EPI == EPI_SIG)  x = 1.0f / (1.0f + __expf(-x));
    if (EPI == EPI_RELU) x = x > 0.0f ? x : 0.0f;
    if (EPI == EPI_RES)  x += resid[idx];
    if (OBF) ((__bf16*)outp)[idx] = f2bf(x);
    else     ((float*)outp)[idx]  = x;
  }
}

// ---------------------------------------------------------------------------
// Tied attention logits: af[h][i][j] = sum_n sum_k q[n,i,h,k]*k2[n,j,h,k]
// One wave per 16x16 (i,j) tile per head; 512 WMMAs accumulated over n.
// ---------------------------------------------------------------------------
__global__ __launch_bounds__(32) void k_logits(const __bf16* __restrict__ q,
                                               const __bf16* __restrict__ k,
                                               float* __restrict__ af) {
  const int it = blockIdx.x, jt = blockIdx.y, h = blockIdx.z;
  const int lane = threadIdx.x;
  v8f acc = {};
  const __bf16* qb = q + (size_t)it * 16 * 128 + h * 32;
  const __bf16* kb = k + (size_t)jt * 16 * 128 + h * 32;
  for (int n = 0; n < 512; ++n) {
    const size_t ro = (size_t)n * (512 * 128);
    __builtin_prefetch(qb + ro + 512 * 128, 0, 3);   // global_prefetch_b8
    __builtin_prefetch(kb + ro + 512 * 128, 0, 3);
    v16bf a = frag_a(qb + ro, 128, lane);
    v16bf b = frag_b(kb + ro, 128, lane);
    acc = wmma_bf16(a, b, acc);
  }
  const int j = jt * 16 + (lane & 15);
  const int half = lane >> 4;
#pragma unroll
  for (int r = 0; r < 8; ++r) {
    const int i = it * 16 + half * 8 + r;
    af[((size_t)h * 512 + i) * 512 + j] = acc[r];
  }
}

// ---------------------------------------------------------------------------
// Softmax over j (with RBF logit bias), f32 -> bf16 probabilities.
// One wave per (i,h); wave32 shuffle reductions.
// ---------------------------------------------------------------------------
__global__ __launch_bounds__(32) void k_softmax(const float* __restrict__ af,
                                                const float* __restrict__ lb,
                                                __bf16* __restrict__ ab) {
  const int i = blockIdx.x, h = blockIdx.y, lane = threadIdx.x;
  const float* row = af + ((size_t)h * 512 + i) * 512;
  float v[16];
  float mx = -3.0e38f;
#pragma unroll
  for (int t = 0; t < 16; ++t) {
    const int j = t * 32 + lane;
    const float x = row[j] + lb[((size_t)i * 512 + j) * 4 + h];
    v[t] = x;
    mx = fmaxf(mx, x);
  }
  mx = wred_max(mx);
  float s = 0.f;
#pragma unroll
  for (int t = 0; t < 16; ++t) { v[t] = __expf(v[t] - mx); s += v[t]; }
  s = wred_sum(s);
  const float inv = 1.0f / s;
  __bf16* orow = ab + ((size_t)h * 512 + i) * 512;
#pragma unroll
  for (int t = 0; t < 16; ++t) orow[t * 32 + lane] = f2bf(v[t] * inv);
}

// ---------------------------------------------------------------------------
// AV: av[kk,i,h,d] = sum_j ab[h][i][j] * v[kk,j,h,d].
// Block = 2 waves (one per 16-wide d tile); K=j accumulated in 32 WMMA steps.
// ---------------------------------------------------------------------------
__global__ __launch_bounds__(64) void k_av(const __bf16* __restrict__ ab,
                                           const __bf16* __restrict__ v,
                                           float* __restrict__ av) {
  const int kk = blockIdx.x, it = blockIdx.y, h = blockIdx.z;
  const int wave = threadIdx.x >> 5;
  const int lane = threadIdx.x & 31;
  v8f acc = {};
  const __bf16* arow = ab + ((size_t)h * 512 + it * 16) * 512;  // [i][j]
#pragma unroll 4
  for (int jt = 0; jt < 32; ++jt) {
    v16bf a = frag_a(arow + jt * 32, 512, lane);
    v16bf b = frag_b_t(v + ((size_t)kk * 512 + jt * 32) * 128 + h * 32 + wave * 16,
                       128, lane);
    acc = wmma_bf16(a, b, acc);
  }
  const int n = lane & 15, half = lane >> 4;
#pragma unroll
  for (int r = 0; r < 8; ++r) {
    const int i = it * 16 + half * 8 + r;
    av[((size_t)kk * 512 + i) * 128 + h * 32 + wave * 16 + n] = acc[r];
  }
}

__global__ __launch_bounds__(256) void k_gatemul(const float* __restrict__ g,
                                                 const float* __restrict__ a,
                                                 __bf16* __restrict__ o) {
  const size_t i = (size_t)blockIdx.x * 256 + threadIdx.x;
  o[i] = f2bf(g[i] * a[i]);
}

// ---------------------------------------------------------------------------
// Workspace layout
// ---------------------------------------------------------------------------
constexpr size_t O_WT   = 0;                          // 256 KB weight pool
constexpr size_t O_XN   = 0x100000;                   // LN'd pair bf16  (64 MB)
constexpr size_t O_NB   = O_XN + LL * 128 * 2;        // LN'd bias bf16  (32 MB)
constexpr size_t O_Q    = O_NB + LL * 64 * 2;         // q bf16          (64 MB)
constexpr size_t O_K    = O_Q + LL * 128 * 2;         // k bf16          (64 MB)
constexpr size_t O_V    = O_K + LL * 128 * 2;         // v bf16          (64 MB)
constexpr size_t O_GATE = O_V + LL * 128 * 2;         // gate f32        (128 MB)
constexpr size_t O_LB   = O_GATE + LL * 128 * 4;      // logit bias f32  (4 MB)
constexpr size_t O_AF   = O_LB + LL * 4 * 4;          // logits f32      (4 MB)
constexpr size_t O_AB   = O_AF + (size_t)4 * LL * 4;  // probs bf16      (2 MB)
constexpr size_t O_AV   = O_AB + (size_t)4 * LL * 2;  // attn out f32    (128 MB)
constexpr size_t O_P1   = O_AV + LL * 128 * 4;        // pair after row  (128 MB)

// Weight-pool slots (bf16 element offsets)
constexpr size_t SL_Q = 0,     SL_K = 16384, SL_V = 32768, SL_G = 49152;
constexpr size_t SL_B = 65536, SL_O = 66560;
constexpr size_t SL_W1 = 0,    SL_W2 = 32768;   // FF reuses the pool

struct WS {
  __bf16 *wt, *xn, *nb, *qb, *kb, *vb, *ab, *gated, *hid;
  float *gate, *lb, *af, *av, *p1;
};
static WS make_ws(char* W) {
  WS w;
  w.wt   = (__bf16*)(W + O_WT);
  w.xn   = (__bf16*)(W + O_XN);
  w.nb   = (__bf16*)(W + O_NB);
  w.qb   = (__bf16*)(W + O_Q);
  w.kb   = (__bf16*)(W + O_K);
  w.vb   = (__bf16*)(W + O_V);
  w.gate = (float*)(W + O_GATE);
  w.lb   = (float*)(W + O_LB);
  w.af   = (float*)(W + O_AF);
  w.ab   = (__bf16*)(W + O_AB);
  w.av   = (float*)(W + O_AV);
  w.p1   = (float*)(W + O_P1);
  w.gated = w.xn;  // alias: xn dead once the 4 projections ran
  w.hid   = w.qb;  // alias: q+k regions (128 MB) dead during the FF stage
  return w;
}

// ---------------------------------------------------------------------------
// One axial attention pass (TR=true for the row pass).
// ---------------------------------------------------------------------------
template <bool TR>
static void attn_pass(void* const* d_in, int base, const float* rbf, const float* Xin,
                      const float* resid, float* pair_out, const WS& w,
                      hipStream_t s) {
  const float* lpg = (const float*)d_in[base + 0];
  const float* lpb = (const float*)d_in[base + 1];
  const float* lbg = (const float*)d_in[base + 2];
  const float* lbb = (const float*)d_in[base + 3];
  const float* wq  = (const float*)d_in[base + 4];
  const float* wk  = (const float*)d_in[base + 5];
  const float* wv  = (const float*)d_in[base + 6];
  const float* wbb = (const float*)d_in[base + 7];
  const float* wg  = (const float*)d_in[base + 8];
  const float* bg  = (const float*)d_in[base + 9];
  const float* wo  = (const float*)d_in[base + 10];
  const float* bo  = (const float*)d_in[base + 11];

  const float SCALE = 0.17677669529663687f;  // 1/sqrt(32)

  k_prep<<<64, 256, 0, s>>>(wq, w.wt + SL_Q, 128, 128, 128, SCALE);
  k_prep<<<64, 256, 0, s>>>(wk, w.wt + SL_K, 128, 128, 128, 1.0f / 512.0f);
  k_prep<<<64, 256, 0, s>>>(wv, w.wt + SL_V, 128, 128, 128, 1.0f);
  k_prep<<<64, 256, 0, s>>>(wg, w.wt + SL_G, 128, 128, 128, 1.0f);
  k_prep<<<4, 256, 0, s>>>(wbb, w.wt + SL_B, 64, 4, 16, 1.0f);
  k_prep<<<64, 256, 0, s>>>(wo, w.wt + SL_O, 128, 128, 128, 1.0f);

  k_ln<128, TR><<<LL / 8, 256, 0, s>>>(Xin, lpg, lpb, w.xn);
  k_ln<64, TR><<<LL / 8, 256, 0, s>>>(rbf, lbg, lbb, w.nb);

  dim3 g128((unsigned)(LL / 16), 2);
  k_gemm<128, 128, 128, EPI_NONE, true, false><<<g128, 128, 0, s>>>(
      w.xn, w.wt + SL_Q, nullptr, nullptr, w.qb);
  k_gemm<128, 128, 128, EPI_NONE, true, false><<<g128, 128, 0, s>>>(
      w.xn, w.wt + SL_K, nullptr, nullptr, w.kb);
  k_gemm<128, 128, 128, EPI_NONE, true, false><<<g128, 128, 0, s>>>(
      w.xn, w.wt + SL_V, nullptr, nullptr, w.vb);
  k_gemm<128, 128, 128, EPI_SIG, false, false><<<g128, 128, 0, s>>>(
      w.xn, w.wt + SL_G, bg, nullptr, w.gate);
  dim3 g16((unsigned)(LL / 16), 1);
  k_gemm<64, 16, 4, EPI_NONE, false, false><<<g16, 128, 0, s>>>(
      w.nb, w.wt + SL_B, nullptr, nullptr, w.lb);

  k_logits<<<dim3(32, 32, 4), 32, 0, s>>>(w.qb, w.kb, w.af);
  k_softmax<<<dim3(512, 4), 32, 0, s>>>(w.af, w.lb, w.ab);
  k_av<<<dim3(512, 32, 4), 64, 0, s>>>(w.ab, w.vb, w.av);
  k_gatemul<<<(unsigned)(LL * 128 / 256), 256, 0, s>>>(w.gate, w.av, w.gated);

  k_gemm<128, 128, 128, EPI_RES, false, TR><<<g128, 128, 0, s>>>(
      w.gated, w.wt + SL_O, bo, resid, pair_out);
}

// ---------------------------------------------------------------------------
// Entry point
// ---------------------------------------------------------------------------
extern "C" void kernel_launch(void* const* d_in, const int* in_sizes, int n_in,
                              void* d_out, int out_size, void* d_ws, size_t ws_size,
                              hipStream_t stream) {
  (void)in_sizes; (void)n_in; (void)out_size; (void)ws_size;
  WS w = make_ws((char*)d_ws);

  const float* pair = (const float*)d_in[0];
  const float* rbf  = (const float*)d_in[1];
  float* out = (float*)d_out;

  // pair += row_attn(pair^T)^T      (writes ws.p1)
  attn_pass<true>(d_in, 2, rbf, pair, pair, w.p1, w, stream);
  // pair += col_attn(pair)          (writes d_out)
  attn_pass<false>(d_in, 14, rbf, w.p1, w.p1, out, w, stream);

  // pair += FF(pair)                (in-place on d_out)
  const float* fg = (const float*)d_in[26];
  const float* fb = (const float*)d_in[27];
  const float* w1 = (const float*)d_in[28];
  const float* b1 = (const float*)d_in[29];
  const float* w2 = (const float*)d_in[30];
  const float* b2 = (const float*)d_in[31];

  k_prep<<<128, 256, 0, stream>>>(w1, w.wt + SL_W1, 128, 256, 256, 1.0f);
  k_prep<<<128, 256, 0, stream>>>(w2, w.wt + SL_W2, 256, 128, 128, 1.0f);
  k_ln<128, false><<<LL / 8, 256, 0, stream>>>(out, fg, fb, w.xn);
  dim3 g256((unsigned)(LL / 16), 4);
  k_gemm<128, 256, 256, EPI_RELU, true, false><<<g256, 128, 0, stream>>>(
      w.xn, w.wt + SL_W1, b1, nullptr, w.hid);
  dim3 g128((unsigned)(LL / 16), 2);
  k_gemm<256, 128, 128, EPI_RES, false, false><<<g128, 128, 0, stream>>>(
      w.hid, w.wt + SL_W2, b2, out, out);
}
